// SRUCell_43095701848352
// MI455X (gfx1250) — compile-verified, matrix operand order checked by reading, not run on hardware
//
#include <hip/hip_runtime.h>
#include <hip/hip_bf16.h>
#include <math.h>

// ---------------------------------------------------------------------------
// SRU cell for MI455X (gfx1250, wave32).
// Phase 1: u = x @ W  (M=2048/chunk, K=2048, N=6144) via bf16x3 split-precision
//          v_wmma_f32_16x16x32_bf16 with LDS staging.
// Phase 2: elementwise recurrence scan, one thread per (b, d) lane.
// L is processed in 8 chunks of 64 so the u slab fits in d_ws (~50.6 MB).
// ---------------------------------------------------------------------------

typedef float   v4f  __attribute__((ext_vector_type(4)));
typedef float   v8f  __attribute__((ext_vector_type(8)));
typedef __bf16  v4bf __attribute__((ext_vector_type(4)));
typedef __bf16  v8bf __attribute__((ext_vector_type(8)));
typedef __bf16  v16bf __attribute__((ext_vector_type(16)));

#define DIM    2048
#define N3     6144
#define BATCH  32
#define LTOT   512
#define LCHUNK 64
#define NCHUNK (LTOT / LCHUNK)
#define MROWS  (LCHUNK * BATCH)   // 2048 rows per chunk GEMM

#define SCALE_X 1.7320508075688772f

// Load one 16x32 bf16 fragment (A-row-major or B-col-major, identical striping)
// from an LDS plane laid out as [row][k] with k-pitch 32 bf16.
// lanes 0-15: K {0..7, 16..23}; lanes 16-31: K {8..15, 24..31}.
__device__ __forceinline__ v16bf load_frag(const __bf16* plane, int row, int laneHi) {
    const v8bf p0 = *(const v8bf*)(plane + row * 32 + laneHi * 8);
    const v8bf p1 = *(const v8bf*)(plane + row * 32 + 16 + laneHi * 8);
    v16bf r;
#pragma unroll
    for (int i = 0; i < 8; ++i) { r[i] = p0[i]; r[8 + i] = p1[i]; }
    return r;
}

__device__ __forceinline__ void split_bf16(float f, __bf16& hi, __bf16& lo) {
    hi = (__bf16)f;
    lo = (__bf16)(f - (float)hi);
}

// ---------------------------------------------------------------------------
// GEMM: U[MROWS x N3] = X[MROWS x DIM] * W[DIM x N3], bf16x3 split precision.
// Block: 256 threads (8 waves). C-tile 128x128. Wave grid 2(M) x 4(N),
// each wave owns 4 M-tiles x 2 N-tiles of 16x16.
// ---------------------------------------------------------------------------
__global__ __launch_bounds__(256)
void sru_gemm(const float* __restrict__ X, const float* __restrict__ W,
              float* __restrict__ U) {
    __shared__ __bf16 As_hi[128 * 32];
    __shared__ __bf16 As_lo[128 * 32];
    __shared__ __bf16 Bs_hi[128 * 32];
    __shared__ __bf16 Bs_lo[128 * 32];

    const int tid    = threadIdx.x;
    const int lane   = tid & 31;
    const int wave   = tid >> 5;
    const int laneLo = lane & 15;
    const int laneHi = lane >> 4;
    const int wM     = wave & 1;        // 0..1 -> M offset wM*64
    const int wN     = wave >> 1;       // 0..3 -> N offset wN*32

    const int blockN = blockIdx.x;      // 0..47
    const int blockM = blockIdx.y;      // 0..15

    // Staging assignments
    const int ar = tid >> 1;            // A row 0..127
    const int ac = (tid & 1) * 16;      // A k-offset 0 or 16
    const int bn = (tid & 31) * 4;      // B n-offset 0..124
    const int bk = (tid >> 5) * 4;      // B k-offset 0..28

    const float* gA = X + (size_t)(blockM * 128 + ar) * DIM + ac;
    const float* gB = W + (size_t)bk * N3 + blockN * 128 + bn;

    v8f acc[4][2];
#pragma unroll
    for (int mt = 0; mt < 4; ++mt)
#pragma unroll
        for (int nt = 0; nt < 2; ++nt) acc[mt][nt] = (v8f)(0.0f);

    // Preload K-slab 0 into registers.
    v4f pa[4], pb[4];
#pragma unroll
    for (int j = 0; j < 4; ++j) pa[j] = *(const v4f*)(gA + j * 4);
#pragma unroll
    for (int j = 0; j < 4; ++j) pb[j] = *(const v4f*)(gB + (size_t)j * N3);

    const int KSTEPS = DIM / 32;        // 64
    for (int kk = 0; kk < KSTEPS; ++kk) {
        __syncthreads();                // previous slab's readers are done

        // ---- stage A (hi/lo split), [row][k] ----
#pragma unroll
        for (int j = 0; j < 4; ++j) {
            v4bf h, l;
#pragma unroll
            for (int e = 0; e < 4; ++e) { __bf16 hb, lb; split_bf16(pa[j][e], hb, lb); h[e] = hb; l[e] = lb; }
            *(v4bf*)(As_hi + ar * 32 + ac + j * 4) = h;
            *(v4bf*)(As_lo + ar * 32 + ac + j * 4) = l;
        }
        // ---- stage B transposed (hi/lo split), [n][k] ----
#pragma unroll
        for (int n = 0; n < 4; ++n) {
            v4bf h, l;
#pragma unroll
            for (int j = 0; j < 4; ++j) { __bf16 hb, lb; split_bf16(pb[j][n], hb, lb); h[j] = hb; l[j] = lb; }
            *(v4bf*)(Bs_hi + (bn + n) * 32 + bk) = h;
            *(v4bf*)(Bs_lo + (bn + n) * 32 + bk) = l;
        }
        __syncthreads();

        // Issue next slab's global loads now; they overlap the WMMA burst.
        if (kk + 1 < KSTEPS) {
            const float* nA = gA + (kk + 1) * 32;
            const float* nB = gB + (size_t)(kk + 1) * 32 * N3;
#pragma unroll
            for (int j = 0; j < 4; ++j) pa[j] = *(const v4f*)(nA + j * 4);
#pragma unroll
            for (int j = 0; j < 4; ++j) pb[j] = *(const v4f*)(nB + (size_t)j * N3);
        }

        // ---- fragments + WMMA: u += Ah*Bh + Ah*Bl + Al*Bh ----
        v16bf Bh[2], Bl[2];
#pragma unroll
        for (int nt = 0; nt < 2; ++nt) {
            Bh[nt] = load_frag(Bs_hi, wN * 32 + nt * 16 + laneLo, laneHi);
            Bl[nt] = load_frag(Bs_lo, wN * 32 + nt * 16 + laneLo, laneHi);
        }
#pragma unroll
        for (int mt = 0; mt < 4; ++mt) {
            v16bf Ah = load_frag(As_hi, wM * 64 + mt * 16 + laneLo, laneHi);
            v16bf Al = load_frag(As_lo, wM * 64 + mt * 16 + laneLo, laneHi);
#pragma unroll
            for (int nt = 0; nt < 2; ++nt) {
                acc[mt][nt] = __builtin_amdgcn_wmma_f32_16x16x32_bf16(
                    false, Ah, false, Bh[nt], (short)0, acc[mt][nt], false, false);
                acc[mt][nt] = __builtin_amdgcn_wmma_f32_16x16x32_bf16(
                    false, Ah, false, Bl[nt], (short)0, acc[mt][nt], false, false);
                acc[mt][nt] = __builtin_amdgcn_wmma_f32_16x16x32_bf16(
                    false, Al, false, Bh[nt], (short)0, acc[mt][nt], false, false);
            }
        }
    }

    // ---- store C: VGPR i holds M = i (lanes 0-15) / M = 8+i (lanes 16-31) ----
    const int mrow0 = blockM * 128 + wM * 64;
    const int ncol0 = blockN * 128 + wN * 32;
#pragma unroll
    for (int mt = 0; mt < 4; ++mt) {
#pragma unroll
        for (int nt = 0; nt < 2; ++nt) {
            const int col = ncol0 + nt * 16 + laneLo;
#pragma unroll
            for (int i = 0; i < 8; ++i) {
                const int row = mrow0 + mt * 16 + laneHi * 8 + i;
                U[(size_t)row * N3 + col] = acc[mt][nt][i];
            }
        }
    }
}

// ---------------------------------------------------------------------------
// Scan: one thread per (b, k). Processes LCHUNK steps of the recurrence.
// u layout per chunk: row = l*BATCH + b, triple (x_tilde, u_f, u_r) at col 3k.
// ---------------------------------------------------------------------------
__global__ __launch_bounds__(256)
void sru_scan(const float* __restrict__ x, const float* __restrict__ u,
              const float* __restrict__ wc, const float* __restrict__ bias,
              const float* __restrict__ c_in, float* __restrict__ c_out,
              float* __restrict__ h_out, float* __restrict__ c_final, int l0) {
    const int t = blockIdx.x * blockDim.x + threadIdx.x;   // 0..65535
    const int b = t >> 11;
    const int k = t & (DIM - 1);

    const float fwc = wc[k],        rwc = wc[DIM + k];
    const float fb  = bias[k],      rb  = bias[DIM + k];

    float c = c_in[t];

    for (int l = 0; l < LCHUNK; ++l) {
        const size_t gidx = ((size_t)(l0 + l) * BATCH + b) * DIM + k;
        const float* up = u + ((size_t)(l * BATCH + b)) * N3 + 3 * k;
        const float xt = up[0];
        const float uf = up[1];
        const float ur = up[2];
        const float xv = x[gidx];

        const float f = 1.0f / (1.0f + __expf(-(uf + c * fwc + fb)));
        const float r = 1.0f / (1.0f + __expf(-(ur + c * rwc + rb)));
        c = (c - xt) * f + xt;
        const float xp = xv * SCALE_X;
        h_out[gidx] = (tanhf(c) - xp) * r + xp;
    }

    c_out[t] = c;
    if (c_final) c_final[t] = c;
}

// ---------------------------------------------------------------------------
extern "C" void kernel_launch(void* const* d_in, const int* in_sizes, int n_in,
                              void* d_out, int out_size, void* d_ws, size_t ws_size,
                              hipStream_t stream) {
    const float* x    = (const float*)d_in[0];   // (512, 32, 2048)
    const float* w    = (const float*)d_in[1];   // (2048, 6144)
    const float* wc   = (const float*)d_in[2];   // (4096,)
    const float* bias = (const float*)d_in[3];   // (4096,)
    const float* c0   = (const float*)d_in[4];   // (32, 2048)

    float* out  = (float*)d_out;                                // h then c_final
    float* u_ws = (float*)d_ws;                                 // chunk u slab
    float* c_ws = (float*)((char*)d_ws + (size_t)MROWS * N3 * sizeof(float));

    const dim3 ggemm(N3 / 128, MROWS / 128);    // (48, 16)
    const int scan_blocks = (BATCH * DIM) / 256;  // 256

    for (int chunk = 0; chunk < NCHUNK; ++chunk) {
        const int l0 = chunk * LCHUNK;
        sru_gemm<<<ggemm, 256, 0, stream>>>(
            x + (size_t)l0 * BATCH * DIM, w, u_ws);

        const float* cin  = (chunk == 0) ? c0 : c_ws;
        float* cfin = (chunk == NCHUNK - 1) ? (out + (size_t)LTOT * BATCH * DIM)
                                            : nullptr;
        sru_scan<<<scan_blocks, 256, 0, stream>>>(
            x, u_ws, wc, bias, cin, c_ws, out, cfin, l0);
    }
}
